// RoPEAttention_39152921870605
// MI455X (gfx1250) — matrix operand order repaired and checked by reading, not compiled, not run on hardware
//
#include <hip/hip_runtime.h>
#include <hip/hip_bf16.h>

typedef __attribute__((ext_vector_type(16))) __bf16 v16bf;
typedef __attribute__((ext_vector_type(8)))  __bf16 v8bf;
typedef __attribute__((ext_vector_type(8)))  float  v8f;

#define DIM_  1024
#define HEADS 16
#define HD    64
#define SLEN  2048
#define BATCH 2
#define NTOK  (BATCH * SLEN)   /* 4096 */
#define QKVN  (3 * DIM_)       /* 3072 */

static __device__ __forceinline__ v16bf make_frag(v8bf lo, v8bf hi) {
  union { v16bf v; v8bf h[2]; } u;
  u.h[0] = lo; u.h[1] = hi;
  return u.v;
}

// A fragment (16x32 bf16, row-major source, stride lda in elements).
// ISA layout: lane m=l&15; lanes<16: elems 0-7 = K0-7, 8-15 = K16-23;
//             lanes>=16: elems 0-7 = K8-15, 8-15 = K24-31.
static __device__ __forceinline__ v16bf load_a(const __bf16* base, int row0,
                                               int k0, int lda, int lane) {
  int m   = lane & 15;
  int klo = (lane < 16) ? 0 : 8;
  const __bf16* p = base + (size_t)(row0 + m) * lda + k0 + klo;
  return make_frag(*(const v8bf*)p, *(const v8bf*)(p + 16));
}

// B fragment (32x16 bf16) from an N-major ("transposed") source Bt[n][k].
// ISA layout: lane n=l&15; lanes<16: elems 0-7 = K0-7, 8-15 = K16-23;
//             lanes>=16: elems 0-7 = K8-15, 8-15 = K24-31.
static __device__ __forceinline__ v16bf load_bt(const __bf16* base, int n0,
                                                int k0, int ldb, int lane) {
  int n  = lane & 15;
  int kb = (lane < 16) ? 0 : 8;
  const __bf16* p = base + (size_t)(n0 + n) * ldb + k0 + kb;
  return make_frag(*(const v8bf*)p, *(const v8bf*)(p + 16));
}

static __device__ __forceinline__ v8f wmma_bf16(v16bf a, v16bf b, v8f c) {
  return __builtin_amdgcn_wmma_f32_16x16x32_bf16(false, a, false, b,
                                                 (short)0, c, false, false);
}

// ---------------- prep kernels ----------------

__global__ void k_f32_to_bf16(const float* __restrict__ in,
                              __bf16* __restrict__ out, int n) {
  int i = (blockIdx.x * blockDim.x + threadIdx.x) * 4;
  if (i + 3 < n) {
    float4 v = *(const float4*)(in + i);
    out[i + 0] = (__bf16)v.x;
    out[i + 1] = (__bf16)v.y;
    out[i + 2] = (__bf16)v.z;
    out[i + 3] = (__bf16)v.w;
  }
}

// wt[n][k] = (bf16) w[k][n];   w is [K][N]
__global__ void k_transpose_bf16(const float* __restrict__ w,
                                 __bf16* __restrict__ wt, int K, int N) {
  int idx = blockIdx.x * blockDim.x + threadIdx.x;
  if (idx < K * N) {
    int n = idx / K;
    int k = idx - n * K;
    wt[(size_t)n * K + k] = (__bf16)w[(size_t)k * N + n];
  }
}

// ---------------- QKV projection: [4096,1024] @ [1024,3072] ----------------
// Writes Qraw/Kraw fp32 [B*H][S][D]; V transposed bf16 [B*H][D][S].

__global__ __launch_bounds__(256) void k_gemm_qkv(
    const __bf16* __restrict__ A,   // x bf16 [4096][1024]
    const __bf16* __restrict__ Bt,  // qkv_w^T bf16 [3072][1024]
    const float* __restrict__ bias, // [3072]
    float* __restrict__ Qraw, float* __restrict__ Kraw,
    __bf16* __restrict__ Vt) {
  int lane = threadIdx.x & 31;
  int wave = threadIdx.x >> 5;
  int wm = wave >> 2;                       // 0..1
  int wn = wave & 3;                        // 0..3
  int row0 = blockIdx.y * 64 + wm * 32;
  int col0 = blockIdx.x * 256 + wn * 64;

  v8f acc[2][4] = {};
  for (int k = 0; k < DIM_; k += 32) {
    v16bf a0 = load_a(A, row0, k, DIM_, lane);
    v16bf a1 = load_a(A, row0 + 16, k, DIM_, lane);
    v16bf b0 = load_bt(Bt, col0, k, DIM_, lane);
    v16bf b1 = load_bt(Bt, col0 + 16, k, DIM_, lane);
    v16bf b2 = load_bt(Bt, col0 + 32, k, DIM_, lane);
    v16bf b3 = load_bt(Bt, col0 + 48, k, DIM_, lane);
    acc[0][0] = wmma_bf16(a0, b0, acc[0][0]);
    acc[0][1] = wmma_bf16(a0, b1, acc[0][1]);
    acc[0][2] = wmma_bf16(a0, b2, acc[0][2]);
    acc[0][3] = wmma_bf16(a0, b3, acc[0][3]);
    acc[1][0] = wmma_bf16(a1, b0, acc[1][0]);
    acc[1][1] = wmma_bf16(a1, b1, acc[1][1]);
    acc[1][2] = wmma_bf16(a1, b2, acc[1][2]);
    acc[1][3] = wmma_bf16(a1, b3, acc[1][3]);
  }

#pragma unroll
  for (int mi = 0; mi < 2; mi++)
#pragma unroll
    for (int ni = 0; ni < 4; ni++)
#pragma unroll
      for (int i = 0; i < 8; i++) {
        int rr = row0 + mi * 16 + i + ((lane < 16) ? 0 : 8);
        int cc = col0 + ni * 16 + (lane & 15);
        float val = acc[mi][ni][i] + bias[cc];
        int t = cc >> 10, rem = cc & 1023;
        int h = rem >> 6, d = rem & 63;
        int b = rr >> 11, s = rr & (SLEN - 1);
        size_t bh = (size_t)(b * HEADS + h);
        if (t == 0)       Qraw[(bh * SLEN + s) * HD + d] = val;
        else if (t == 1)  Kraw[(bh * SLEN + s) * HD + d] = val;
        else              Vt[(bh * HD + d) * SLEN + s] = (__bf16)val;
      }
}

// ---------------- RoPE (fp32 rotate -> bf16) ----------------

__global__ void k_rope(const float* __restrict__ Qraw,
                       const float* __restrict__ Kraw,
                       __bf16* __restrict__ Qb, __bf16* __restrict__ Kb) {
  int tid = blockIdx.x * blockDim.x + threadIdx.x;  // BH*S*32 threads
  int j  = tid & 31;
  int s  = (tid >> 5) & (SLEN - 1);
  int bh = tid >> 16;
  size_t base = ((size_t)bh * SLEN + s) * HD + 2 * j;
  float inv = __expf(-(float)j * 0.28782313662425574f);  // ln(1e4)/32
  float ang = (float)s * inv;
  float sn = sinf(ang), cs = cosf(ang);
  float q0 = Qraw[base], q1 = Qraw[base + 1];
  Qb[base]     = (__bf16)(q0 * cs - q1 * sn);
  Qb[base + 1] = (__bf16)(q1 * cs + q0 * sn);
  float k0 = Kraw[base], k1 = Kraw[base + 1];
  Kb[base]     = (__bf16)(k0 * cs - k1 * sn);
  Kb[base + 1] = (__bf16)(k1 * cs + k0 * sn);
}

// ---------------- flash attention (16 query rows / wave) ----------------

__global__ __launch_bounds__(128) void k_flash(
    const __bf16* __restrict__ Q,   // [B*H][S][D]
    const __bf16* __restrict__ K,   // [B*H][S][D]
    const __bf16* __restrict__ Vt,  // [B*H][D][S]
    __bf16* __restrict__ ctx) {     // [B][S][H][D]
  __shared__ __align__(16) __bf16 lds_p[4][16][32];
  int lane = threadIdx.x & 31;
  int wave = threadIdx.x >> 5;
  int qb = blockIdx.x & 31;   // S/64 tiles
  int bh = blockIdx.x >> 5;   // 0..31
  int q0 = qb * 64 + wave * 16;

  const __bf16* Qp = Q  + (size_t)bh * SLEN * HD;
  const __bf16* Kp = K  + (size_t)bh * SLEN * HD;
  const __bf16* Vp = Vt + (size_t)bh * HD * SLEN;

  v16bf aq0 = load_a(Qp, q0, 0, HD, lane);
  v16bf aq1 = load_a(Qp, q0, 32, HD, lane);

  v8f o[4] = {};
  float mrow[8], lrow[8];
#pragma unroll
  for (int i = 0; i < 8; i++) { mrow[i] = -1e30f; lrow[i] = 0.f; }

  int m   = lane & 15;
  int klo = (lane < 16) ? 0 : 8;

  for (int j = 0; j < SLEN; j += 32) {
    v8f s0 = {}, s1 = {};
    v16bf bk;
    bk = load_bt(Kp, j,      0,  HD, lane); s0 = wmma_bf16(aq0, bk, s0);
    bk = load_bt(Kp, j,      32, HD, lane); s0 = wmma_bf16(aq1, bk, s0);
    bk = load_bt(Kp, j + 16, 0,  HD, lane); s1 = wmma_bf16(aq0, bk, s1);
    bk = load_bt(Kp, j + 16, 32, HD, lane); s1 = wmma_bf16(aq1, bk, s1);

#pragma unroll
    for (int i = 0; i < 8; i++) {
      float v0 = s0[i] * 0.125f, v1 = s1[i] * 0.125f;  // 1/sqrt(64)
      float mx = fmaxf(v0, v1);
      mx = fmaxf(mx, __shfl_xor(mx, 1));
      mx = fmaxf(mx, __shfl_xor(mx, 2));
      mx = fmaxf(mx, __shfl_xor(mx, 4));
      mx = fmaxf(mx, __shfl_xor(mx, 8));
      float newm  = fmaxf(mrow[i], mx);
      float alpha = __expf(mrow[i] - newm);
      mrow[i] = newm;
      __bf16 p0 = (__bf16)__expf(v0 - newm);
      __bf16 p1 = (__bf16)__expf(v1 - newm);
      float rs = (float)p0 + (float)p1;
      rs += __shfl_xor(rs, 1);
      rs += __shfl_xor(rs, 2);
      rs += __shfl_xor(rs, 4);
      rs += __shfl_xor(rs, 8);
      lrow[i] = lrow[i] * alpha + rs;
      o[0][i] *= alpha; o[1][i] *= alpha; o[2][i] *= alpha; o[3][i] *= alpha;
      int r = i + ((lane < 16) ? 0 : 8);
      lds_p[wave][r][lane & 15]        = p0;
      lds_p[wave][r][(lane & 15) + 16] = p1;
    }
    __syncthreads();
    const __bf16* pp = &lds_p[wave][m][klo];
    v16bf ap = make_frag(*(const v8bf*)pp, *(const v8bf*)(pp + 16));
#pragma unroll
    for (int dt = 0; dt < 4; dt++) {
      v16bf bv = load_bt(Vp, dt * 16, j, SLEN, lane);
      o[dt] = wmma_bf16(ap, bv, o[dt]);
    }
  }

  int bB = bh >> 4, h = bh & 15;
#pragma unroll
  for (int dt = 0; dt < 4; dt++)
#pragma unroll
    for (int i = 0; i < 8; i++) {
      int s = q0 + i + ((lane < 16) ? 0 : 8);
      float val = o[dt][i] / lrow[i];
      size_t idx = (((size_t)bB * SLEN + s) * HEADS + h) * HD + dt * 16 + (lane & 15);
      ctx[idx] = (__bf16)val;
    }
}

// ---------------- output projection: [4096,1024] @ [1024,1024] ----------------

__global__ __launch_bounds__(256) void k_gemm_out(
    const __bf16* __restrict__ A,   // ctx bf16 [4096][1024]
    const __bf16* __restrict__ Bt,  // out_w^T bf16 [1024][1024]
    const float* __restrict__ bias, // [1024]
    float* __restrict__ C) {        // [4096][1024]
  int lane = threadIdx.x & 31;
  int wave = threadIdx.x >> 5;
  int wm = wave >> 2;
  int wn = wave & 3;
  int row0 = blockIdx.y * 64 + wm * 32;
  int col0 = blockIdx.x * 256 + wn * 64;

  v8f acc[2][4] = {};
  for (int k = 0; k < DIM_; k += 32) {
    v16bf a0 = load_a(A, row0, k, DIM_, lane);
    v16bf a1 = load_a(A, row0 + 16, k, DIM_, lane);
    v16bf b0 = load_bt(Bt, col0, k, DIM_, lane);
    v16bf b1 = load_bt(Bt, col0 + 16, k, DIM_, lane);
    v16bf b2 = load_bt(Bt, col0 + 32, k, DIM_, lane);
    v16bf b3 = load_bt(Bt, col0 + 48, k, DIM_, lane);
    acc[0][0] = wmma_bf16(a0, b0, acc[0][0]);
    acc[0][1] = wmma_bf16(a0, b1, acc[0][1]);
    acc[0][2] = wmma_bf16(a0, b2, acc[0][2]);
    acc[0][3] = wmma_bf16(a0, b3, acc[0][3]);
    acc[1][0] = wmma_bf16(a1, b0, acc[1][0]);
    acc[1][1] = wmma_bf16(a1, b1, acc[1][1]);
    acc[1][2] = wmma_bf16(a1, b2, acc[1][2]);
    acc[1][3] = wmma_bf16(a1, b3, acc[1][3]);
  }

#pragma unroll
  for (int mi = 0; mi < 2; mi++)
#pragma unroll
    for (int ni = 0; ni < 4; ni++)
#pragma unroll
      for (int i = 0; i < 8; i++) {
        int rr = row0 + mi * 16 + i + ((lane < 16) ? 0 : 8);
        int cc = col0 + ni * 16 + (lane & 15);
        C[(size_t)rr * DIM_ + cc] = acc[mi][ni][i] + bias[cc];
      }
}

// ---------------- host launcher ----------------

extern "C" void kernel_launch(void* const* d_in, const int* in_sizes, int n_in,
                              void* d_out, int out_size, void* d_ws,
                              size_t ws_size, hipStream_t stream) {
  (void)in_sizes; (void)n_in; (void)out_size; (void)ws_size;
  const float* x     = (const float*)d_in[0];
  const float* qkv_w = (const float*)d_in[1];
  const float* qkv_b = (const float*)d_in[2];
  const float* out_w = (const float*)d_in[3];
  const float* out_b = (const float*)d_in[4];

  char* ws = (char*)d_ws;
  size_t off = 0;
  auto alloc = [&](size_t bytes) -> void* {
    void* p = ws + off;
    off += (bytes + 255) & ~(size_t)255;
    return p;
  };
  const size_t NE = (size_t)NTOK * DIM_;             // 4,194,304
  __bf16* x_bf  = (__bf16*)alloc(NE * 2);
  __bf16* qkvwT = (__bf16*)alloc((size_t)QKVN * DIM_ * 2);
  __bf16* outwT = (__bf16*)alloc((size_t)DIM_ * DIM_ * 2);
  float*  Qraw  = (float*)alloc(NE * 4);
  float*  Kraw  = (float*)alloc(NE * 4);
  __bf16* Qb    = (__bf16*)alloc(NE * 2);
  __bf16* Kb    = (__bf16*)alloc(NE * 2);
  __bf16* Vt    = (__bf16*)alloc(NE * 2);
  __bf16* ctx   = (__bf16*)alloc(NE * 2);

  k_f32_to_bf16<<<(int)(NE / 1024), 256, 0, stream>>>(x, x_bf, (int)NE);
  k_transpose_bf16<<<(QKVN * DIM_ + 255) / 256, 256, 0, stream>>>(qkv_w, qkvwT,
                                                                  DIM_, QKVN);
  k_transpose_bf16<<<(DIM_ * DIM_ + 255) / 256, 256, 0, stream>>>(out_w, outwT,
                                                                  DIM_, DIM_);
  k_gemm_qkv<<<dim3(QKVN / 256, NTOK / 64), 256, 0, stream>>>(
      x_bf, qkvwT, qkv_b, Qraw, Kraw, Vt);
  k_rope<<<(BATCH * HEADS * SLEN * 32) / 256, 256, 0, stream>>>(Qraw, Kraw, Qb,
                                                                Kb);
  k_flash<<<BATCH * HEADS * (SLEN / 64), 128, 0, stream>>>(Qb, Kb, Vt, ctx);
  k_gemm_out<<<dim3(DIM_ / 256, NTOK / 64), 256, 0, stream>>>(
      ctx, outwT, out_b, (float*)d_out);
}
